// Attention_86079734546756
// MI455X (gfx1250) — matrix operand order repaired and checked by reading, not compiled, not run on hardware
//
#include <hip/hip_runtime.h>
#include <hip/hip_bf16.h>

// ---------------------------------------------------------------------------
// MHA block for MI455X (gfx1250, wave32, WMMA).
// bf16 WMMA everywhere; LDS double-buffered staging with CDNA5 async
// global->LDS copies; flash attention with block-shared K/V tiles.
// ---------------------------------------------------------------------------

typedef __attribute__((ext_vector_type(16))) __bf16 v16bf;
typedef __attribute__((ext_vector_type(8)))  float  v8f;
typedef __attribute__((ext_vector_type(4)))  int    v4i;

#define B_DIM  4
#define S_DIM  2048
#define D_DIM  1024
#define H_DIM  16
#define DH_DIM 64
#define NEG_BIG (-100000.0f)

#if defined(__has_builtin)
#if __has_builtin(__builtin_amdgcn_global_load_async_to_lds_b128)
#define USE_ASYNC_LDS 1
#endif
#if __has_builtin(__builtin_amdgcn_s_wait_asynccnt)
#define HAVE_WAIT_ASYNC 1
#endif
#endif

static __device__ __forceinline__ unsigned short f2bf(float f) {
    unsigned int u = __float_as_uint(f);
    unsigned int r = u + 0x7FFFu + ((u >> 16) & 1u);   // round-to-nearest-even
    return (unsigned short)(r >> 16);
}

// Copy 16 bytes global -> LDS (async on CDNA5, ASYNCcnt-tracked).
static __device__ __forceinline__ void cp16(unsigned short* lds, const unsigned short* g) {
#if defined(USE_ASYNC_LDS)
    __builtin_amdgcn_global_load_async_to_lds_b128(
        (__attribute__((address_space(1))) v4i*)g,
        (__attribute__((address_space(3))) v4i*)lds, 0, 0);
#else
    *(uint4*)lds = *(const uint4*)g;
#endif
}

static __device__ __forceinline__ void wait_stage() {
#if defined(USE_ASYNC_LDS) && defined(HAVE_WAIT_ASYNC)
    __builtin_amdgcn_s_wait_asynccnt(0);
#endif
    __syncthreads();
}

union FragB16 {
    v16bf v;
    unsigned int d[8];
};

// A fragment: 16x32 bf16, row-major source, leading dim lda (elements).
// Lanes 0-15 hold M=0..15 (lanes 16-31 same M, other K half);
// VGPR j<4 -> K = 2j + 8*half, j>=4 -> K = 16 + 2(j-4) + 8*half.
static __device__ __forceinline__ v16bf load_a_frag(const unsigned short* A, int lda, int lane) {
    int m  = lane & 15;
    int kh = (lane >> 4) & 1;
    FragB16 f;
#pragma unroll
    for (int j = 0; j < 8; ++j) {
        int kk = ((j < 4) ? (2 * j) : (16 + 2 * (j - 4))) + kh * 8;
        f.d[j] = *(const unsigned int*)(A + (size_t)m * lda + kk);
    }
    return f.v;
}

// B fragment: 32x16 bf16 (KxN), source row-major [K][N], leading dim ldb.
// Lane = N column (lane & 15); VGPR j holds K = k0, k0+1 in the same K order as A.
static __device__ __forceinline__ v16bf load_b_frag(const unsigned short* Bt, int ldb, int lane) {
    int n  = lane & 15;
    int kh = (lane >> 4) & 1;
    FragB16 f;
#pragma unroll
    for (int j = 0; j < 8; ++j) {
        int kk = ((j < 4) ? (2 * j) : (16 + 2 * (j - 4))) + kh * 8;
        f.d[j] = (unsigned int)Bt[(size_t)kk * ldb + n] | ((unsigned int)Bt[(size_t)(kk + 1) * ldb + n] << 16);
    }
    return f.v;
}

static __device__ __forceinline__ v8f wmma_bf16(v16bf a, v16bf b, v8f c) {
    return __builtin_amdgcn_wmma_f32_16x16x32_bf16(
        false, a, false, b, (short)0, c, false, false);
}

// ---------------------------------------------------------------------------
// Elementwise fp32 -> bf16
// ---------------------------------------------------------------------------
__global__ void f32_to_bf16_kernel(const float* __restrict__ in,
                                   unsigned short* __restrict__ out, long n) {
    long i = (long)blockIdx.x * blockDim.x + threadIdx.x;
    if (i < n) out[i] = f2bf(in[i]);
}

// W[N][K] fp32 -> Wt[K][N] bf16
__global__ void w_transpose_bf16_kernel(const float* __restrict__ W,
                                        unsigned short* __restrict__ Wt,
                                        int N, int K) {
    long i = (long)blockIdx.x * blockDim.x + threadIdx.x;
    if (i < (long)N * K) {
        int n = (int)(i / K);
        int k = (int)(i % K);
        Wt[(size_t)k * N + n] = f2bf(W[i]);
    }
}

// ---------------------------------------------------------------------------
// WMMA GEMM with double-buffered async LDS staging.
// C[M,N] = A[M,K](bf16) * Wt[K,N](bf16) + bias[N]
// mode 0: fp32 out [M,N];  mode 1: bf16 [B,H,S,DH];  mode 2: bf16 [B,H,DH,S]
// block = 128 threads (4 waves); block tile 64x64; wave tile 16x64; K-step 32.
// ---------------------------------------------------------------------------
__global__ void gemm_bf16_kernel(const unsigned short* __restrict__ A,
                                 const unsigned short* __restrict__ Bt,
                                 const float* __restrict__ bias,
                                 void* __restrict__ out,
                                 int M, int N, int K, int mode) {
    __shared__ __align__(16) unsigned short a_sh[2][64 * 32];   // [m][k]
    __shared__ __align__(16) unsigned short b_sh[2][32 * 64];   // [k][n]

    int tid  = threadIdx.x;
    int lane = tid & 31;
    int wave = tid >> 5;
    int mblk = blockIdx.x * 64;
    int nblk = blockIdx.y * 64;

    auto stage = [&](int buf, int kb) {
        // A tile: 64 rows x 32 cols (64B/row): thread -> row tid>>1, 2x16B
        int r = tid >> 1, hf = tid & 1;
        const unsigned short* ga = A + (size_t)(mblk + r) * K + kb + hf * 16;
        cp16(&a_sh[buf][r * 32 + hf * 16],     ga);
        cp16(&a_sh[buf][r * 32 + hf * 16 + 8], ga + 8);
        // B tile: 32 rows x 64 cols (128B/row): thread -> row tid>>2, 2x16B
        int r2 = tid >> 2, c = tid & 3;
        const unsigned short* gb = Bt + (size_t)(kb + r2) * N + nblk + c * 16;
        cp16(&b_sh[buf][r2 * 64 + c * 16],     gb);
        cp16(&b_sh[buf][r2 * 64 + c * 16 + 8], gb + 8);
    };

    v8f acc[4];
#pragma unroll
    for (int t = 0; t < 4; ++t)
#pragma unroll
        for (int r = 0; r < 8; ++r) acc[t][r] = 0.0f;

    stage(0, 0);
    int buf = 0;
    for (int kb = 0; kb < K; kb += 32) {
        wait_stage();                        // staged tile [buf] ready, all waves past reads
        if (kb + 32 < K) stage(buf ^ 1, kb + 32);   // overlap copy with WMMA
        v16bf a = load_a_frag(&a_sh[buf][wave * 16 * 32], 32, lane);
#pragma unroll
        for (int t = 0; t < 4; ++t) {
            v16bf b = load_b_frag(&b_sh[buf][t * 16], 64, lane);
            acc[t] = wmma_bf16(a, b, acc[t]);
        }
        buf ^= 1;
    }

    int half = lane >> 4;
    int nl   = lane & 15;
    int mrow = mblk + wave * 16;
#pragma unroll
    for (int t = 0; t < 4; ++t) {
        int col = nblk + t * 16 + nl;
        float bs = bias[col];
#pragma unroll
        for (int r = 0; r < 8; ++r) {
            int row = mrow + r + 8 * half;
            float v = acc[t][r] + bs;
            if (mode == 0) {
                ((float*)out)[(size_t)row * N + col] = v;
            } else {
                int b  = row / S_DIM, s  = row % S_DIM;
                int h  = col / DH_DIM, dh = col % DH_DIM;
                if (mode == 1)
                    ((unsigned short*)out)[(((size_t)(b * H_DIM + h)) * S_DIM + s) * DH_DIM + dh] = f2bf(v);
                else
                    ((unsigned short*)out)[(((size_t)(b * H_DIM + h)) * DH_DIM + dh) * S_DIM + s] = f2bf(v);
            }
        }
    }
}

// ---------------------------------------------------------------------------
// Flash attention: 4 waves/block, all sharing one (b,h); each wave owns a
// 16-row Q block; K/V tiles (32 keys) staged in LDS once per block,
// double-buffered via async copies. Online softmax; P re-laid out in LDS.
// ---------------------------------------------------------------------------
__global__ void flash_attn_kernel(const unsigned short* __restrict__ Qh,  // [B,H,S,DH]
                                  const unsigned short* __restrict__ Kt,  // [B,H,DH,S]
                                  const unsigned short* __restrict__ Vh,  // [B,H,S,DH]
                                  const unsigned char* __restrict__ mask, // [B,1,S,S]
                                  unsigned short* __restrict__ Oo) {      // [B,S,D] bf16
    __shared__ __align__(16) unsigned short kt_sh[2][64 * 32]; // [dh][kpos]
    __shared__ __align__(16) unsigned short vt_sh[2][32 * 64]; // [kpos][dh]
    __shared__ __align__(16) unsigned short psh[4][16 * 32];   // per-wave P

    int tid  = threadIdx.x;
    int lane = tid & 31;
    int wave = tid >> 5;
    int id0  = blockIdx.x * 4;                 // 4 q-blocks of same (b,h)
    int tmp  = id0 / (S_DIM / 16);
    int h    = tmp % H_DIM;
    int b    = tmp / H_DIM;
    int qb   = (id0 + wave) % (S_DIM / 16);

    const unsigned short* Q  = Qh + (((size_t)(b * H_DIM + h)) * S_DIM + (size_t)qb * 16) * DH_DIM;
    const unsigned short* KT = Kt + ((size_t)(b * H_DIM + h)) * DH_DIM * S_DIM;
    const unsigned short* V  = Vh + ((size_t)(b * H_DIM + h)) * S_DIM * DH_DIM;
    const unsigned char*  Mr = mask + (size_t)b * S_DIM * S_DIM + (size_t)qb * 16 * S_DIM;

    auto stage_kv = [&](int buf, int kb) {
        // K^T tile: 64 dh-rows x 32 keys (64B/row)
        int r = tid >> 1, hf = tid & 1;
        const unsigned short* gk = KT + (size_t)r * S_DIM + kb + hf * 16;
        cp16(&kt_sh[buf][r * 32 + hf * 16],     gk);
        cp16(&kt_sh[buf][r * 32 + hf * 16 + 8], gk + 8);
        // V tile: 32 key-rows x 64 dh (128B/row)
        int r2 = tid >> 2, c = tid & 3;
        const unsigned short* gv = V + (size_t)(kb + r2) * DH_DIM + c * 16;
        cp16(&vt_sh[buf][r2 * 64 + c * 16],     gv);
        cp16(&vt_sh[buf][r2 * 64 + c * 16 + 8], gv + 8);
    };

    // Loop-invariant Q fragments (two K-steps over DH=64).
    v16bf qa0 = load_a_frag(Q, DH_DIM, lane);
    v16bf qa1 = load_a_frag(Q + 32, DH_DIM, lane);

    float m_run[8], l_run[8];
    v8f o[4];
#pragma unroll
    for (int r = 0; r < 8; ++r) { m_run[r] = -3.0e30f; l_run[r] = 0.0f; }
#pragma unroll
    for (int t = 0; t < 4; ++t)
#pragma unroll
        for (int r = 0; r < 8; ++r) o[t][r] = 0.0f;

    int half = lane >> 4;
    int nl   = lane & 15;

    stage_kv(0, 0);
    int buf = 0;
    for (int kb = 0; kb < S_DIM; kb += 32) {
        wait_stage();
        if (kb + 32 < S_DIM) stage_kv(buf ^ 1, kb + 32);

        // ---- scores: 16x32 = Q(16x64) @ K^T(64x32), two N-tiles ----
        v8f s0, s1;
#pragma unroll
        for (int r = 0; r < 8; ++r) { s0[r] = 0.0f; s1[r] = 0.0f; }
        s0 = wmma_bf16(qa0, load_b_frag(&kt_sh[buf][0],            32, lane), s0);
        s0 = wmma_bf16(qa1, load_b_frag(&kt_sh[buf][32 * 32],      32, lane), s0);
        s1 = wmma_bf16(qa0, load_b_frag(&kt_sh[buf][16],           32, lane), s1);
        s1 = wmma_bf16(qa1, load_b_frag(&kt_sh[buf][32 * 32 + 16], 32, lane), s1);

        // ---- mask + online softmax (rows live in 16-lane halves) ----
#pragma unroll
        for (int r = 0; r < 8; ++r) {
            int qrow = r + 8 * half;
            const unsigned char* mp = Mr + (size_t)qrow * S_DIM + kb;
            if (mp[nl])      s0[r] = NEG_BIG;
            if (mp[16 + nl]) s1[r] = NEG_BIG;

            float rmax = fmaxf(s0[r], s1[r]);
#pragma unroll
            for (int off = 1; off < 16; off <<= 1)
                rmax = fmaxf(rmax, __shfl_xor(rmax, off, 32));

            float mnew = fmaxf(m_run[r], rmax);
            float p0 = __expf(s0[r] - mnew);
            float p1 = __expf(s1[r] - mnew);
            float rsum = p0 + p1;
#pragma unroll
            for (int off = 1; off < 16; off <<= 1)
                rsum += __shfl_xor(rsum, off, 32);

            float sc = __expf(m_run[r] - mnew);
            l_run[r] = l_run[r] * sc + rsum;
            m_run[r] = mnew;
#pragma unroll
            for (int t = 0; t < 4; ++t) o[t][r] *= sc;

            psh[wave][qrow * 32 + nl]      = f2bf(p0);
            psh[wave][qrow * 32 + 16 + nl] = f2bf(p1);
        }
        __threadfence_block();                       // C-layout -> A-layout
        v16bf pa = load_a_frag(&psh[wave][0], 32, lane);
        __threadfence_block();

        // ---- O(16x64) += P(16x32) @ V(32x64) ----
#pragma unroll
        for (int t = 0; t < 4; ++t) {
            v16bf vb = load_b_frag(&vt_sh[buf][t * 16], 64, lane);
            o[t] = wmma_bf16(pa, vb, o[t]);
        }
        buf ^= 1;
    }

    // ---- normalize and store bf16 [B,S,D] ----
#pragma unroll
    for (int t = 0; t < 4; ++t) {
        int col = h * DH_DIM + t * 16 + nl;
#pragma unroll
        for (int r = 0; r < 8; ++r) {
            int srow = qb * 16 + r + 8 * half;
            float inv = (l_run[r] > 0.0f) ? (1.0f / l_run[r]) : 0.0f;
            Oo[((size_t)b * S_DIM + srow) * D_DIM + col] = f2bf(o[t][r] * inv);
        }
    }
}

// ---------------------------------------------------------------------------
// Host-side launch
// ---------------------------------------------------------------------------
extern "C" void kernel_launch(void* const* d_in, const int* in_sizes, int n_in,
                              void* d_out, int out_size, void* d_ws, size_t ws_size,
                              hipStream_t stream) {
    const int M = B_DIM * S_DIM;          // 8192
    const int N = D_DIM;                  // 1024
    const int K = D_DIM;                  // 1024
    const long ACT = (long)M * K;
    const long WEL = (long)N * K;

    const float* q    = (const float*)d_in[0];
    const float* k    = (const float*)d_in[1];
    const float* v    = (const float*)d_in[2];
    const unsigned char* mask = (const unsigned char*)d_in[3];
    const float* wq_w = (const float*)d_in[4];
    const float* wq_b = (const float*)d_in[5];
    const float* wk_w = (const float*)d_in[6];
    const float* wk_b = (const float*)d_in[7];
    const float* wv_w = (const float*)d_in[8];
    const float* wv_b = (const float*)d_in[9];
    const float* wo_w = (const float*)d_in[10];
    const float* wo_b = (const float*)d_in[11];

    char* ws = (char*)d_ws;
    size_t off = 0;
    auto carve = [&](size_t bytes) { char* p = ws + off; off = (off + bytes + 255) & ~(size_t)255; return p; };
    unsigned short* wqt = (unsigned short*)carve(WEL * 2);
    unsigned short* wkt = (unsigned short*)carve(WEL * 2);
    unsigned short* wvt = (unsigned short*)carve(WEL * 2);
    unsigned short* wot = (unsigned short*)carve(WEL * 2);
    unsigned short* xq  = (unsigned short*)carve(ACT * 2);
    unsigned short* xk  = (unsigned short*)carve(ACT * 2);
    unsigned short* xv  = (unsigned short*)carve(ACT * 2);
    unsigned short* qh  = (unsigned short*)carve(ACT * 2);  // [B,H,S,DH]
    unsigned short* kt  = (unsigned short*)carve(ACT * 2);  // [B,H,DH,S]
    unsigned short* vh  = (unsigned short*)carve(ACT * 2);  // [B,H,S,DH]
    unsigned short* ao  = (unsigned short*)carve(ACT * 2);  // [B,S,D]

    {
        int thr = 256;
        int gA = (int)((ACT + thr - 1) / thr);
        f32_to_bf16_kernel<<<gA, thr, 0, stream>>>(q, xq, ACT);
        f32_to_bf16_kernel<<<gA, thr, 0, stream>>>(k, xk, ACT);
        f32_to_bf16_kernel<<<gA, thr, 0, stream>>>(v, xv, ACT);
        int gW = (int)((WEL + thr - 1) / thr);
        w_transpose_bf16_kernel<<<gW, thr, 0, stream>>>(wq_w, wqt, N, K);
        w_transpose_bf16_kernel<<<gW, thr, 0, stream>>>(wk_w, wkt, N, K);
        w_transpose_bf16_kernel<<<gW, thr, 0, stream>>>(wv_w, wvt, N, K);
        w_transpose_bf16_kernel<<<gW, thr, 0, stream>>>(wo_w, wot, N, K);
    }

    dim3 ggrid(M / 64, N / 64);
    dim3 gblk(128);
    gemm_bf16_kernel<<<ggrid, gblk, 0, stream>>>(xq, wqt, wq_b, qh, M, N, K, 1);
    gemm_bf16_kernel<<<ggrid, gblk, 0, stream>>>(xk, wkt, wk_b, kt, M, N, K, 2);
    gemm_bf16_kernel<<<ggrid, gblk, 0, stream>>>(xv, wvt, wv_b, vh, M, N, K, 1);

    int qblocks = B_DIM * H_DIM * (S_DIM / 16);
    flash_attn_kernel<<<qblocks / 4, 128, 0, stream>>>(qh, kt, vh, mask, ao);

    gemm_bf16_kernel<<<ggrid, gblk, 0, stream>>>(ao, wot, wo_b, (float*)d_out, M, N, K, 0);

    (void)in_sizes; (void)n_in; (void)out_size; (void)ws_size;
}
